// MaskAttention_20804821582358
// MI455X (gfx1250) — matrix operand order repaired and checked by reading, not compiled, not run on hardware
//
#include <hip/hip_runtime.h>
#include <hip/hip_bf16.h>

// ---------------- types & constants ----------------
typedef __bf16 bf16_t;
typedef __attribute__((ext_vector_type(16))) __bf16 v16bf;
typedef __attribute__((ext_vector_type(8)))  float  v8f;

#define HEADS   8
#define DH      64
#define NTOK    1024
#define CIN     256
#define QKVO    1536
#define VDIM    512
#define OUTCH   256
#define NBATCH  16

// softmax scale (dh^-0.5 = 0.125) folded together with log2(e): attention uses exp2
#define QSCALE  0.18033688011112042f

union BF16Frag {
    v16bf v;
    uint4 u4[2];
    unsigned short h[16];
};

__device__ __forceinline__ unsigned short bfbits(float x) {
    union { __bf16 b; unsigned short s; } cv;
    cv.b = (__bf16)x;
    return cv.s;
}

__device__ __forceinline__ float fast_exp2(float x) {
    return __builtin_amdgcn_exp2f(x);      // v_exp_f32 (native exp2)
}

__device__ __forceinline__ v8f wmma_bf16(v16bf a, v16bf b, v8f c) {
    return __builtin_amdgcn_wmma_f32_16x16x32_bf16(false, a, false, b, (short)0, c, false, false);
}

// ---- gfx1250 async global->LDS copies (ASYNCcnt) ----
__device__ __forceinline__ void async_copy_b32(unsigned lds_off, const void* gaddr) {
    asm volatile("global_load_async_to_lds_b32 %0, %1, off"
                 :: "v"(lds_off), "v"((unsigned long long)(size_t)gaddr)
                 : "memory");
}
__device__ __forceinline__ void async_copy_b128(unsigned lds_off, const void* gaddr) {
    asm volatile("global_load_async_to_lds_b128 %0, %1, off"
                 :: "v"(lds_off), "v"((unsigned long long)(size_t)gaddr)
                 : "memory");
}
__device__ __forceinline__ void wait_async0() {
    asm volatile("s_wait_asynccnt 0" ::: "memory");
}

// A fragment (16x32) from bf16 row-major matrix W[ld].
__device__ __forceinline__ v16bf load_a_bf16_w(const bf16_t* __restrict__ W, int ld,
                                               int rowbase, int colbase, int lane) {
    int m   = lane & 15;
    int off = (lane >> 4) << 4;
    const bf16_t* p = W + (size_t)(rowbase + m) * ld + colbase + off;
    BF16Frag f;
    f.u4[0] = *(const uint4*)(const void*)(p);
    f.u4[1] = *(const uint4*)(const void*)(p + 8);
    return f.v;
}

// A fragment (16x32) from bf16 memory; caller passes &row[off].
__device__ __forceinline__ v16bf load_a_bf16(const bf16_t* __restrict__ p) {
    BF16Frag f;
    f.u4[0] = *(const uint4*)(const void*)(p);
    f.u4[1] = *(const uint4*)(const void*)(p + 8);
    return f.v;
}

// B fragment (32x16): pointer to this lane's 16 contiguous bf16 K-values.
__device__ __forceinline__ v16bf load_b_bf16(const bf16_t* __restrict__ p) {
    BF16Frag f;
    f.u4[0] = *(const uint4*)(const void*)(p);
    f.u4[1] = *(const uint4*)(const void*)(p + 8);
    return f.v;
}

#define XS_STRIDE 36   // padded fp32 LDS row stride (dwords)
#define BS_STRIDE 72   // padded bf16 LDS row stride (halves) for 64-wide K tiles

// ---------------- prep A: elementwise fp32 -> bf16 (weights) ----------------
__global__ __launch_bounds__(256) void cvt_kernel(const float* __restrict__ src,
                                                  bf16_t* __restrict__ dst, int nelem) {
    int idx = (blockIdx.x * 256 + threadIdx.x) * 8;
    if (idx + 8 > nelem) return;
    union { float4 v; float f[4]; } a0, a1;
    a0.v = *(const float4*)(const void*)(src + idx);
    a1.v = *(const float4*)(const void*)(src + idx + 4);
    union { uint4 u; unsigned short s[8]; } pk;
#pragma unroll
    for (int i = 0; i < 4; ++i) pk.s[i]     = bfbits(a0.f[i]);
#pragma unroll
    for (int i = 0; i < 4; ++i) pk.s[4 + i] = bfbits(a1.f[i]);
    *(uint4*)(void*)(dst + idx) = pk.u;
}

// ---------------- prep B: x[b][c][tok] -> XB[b][tok][c] bf16 ----------------
__global__ __launch_bounds__(256) void xprep_kernel(const float* __restrict__ x,
                                                    bf16_t* __restrict__ XB) {
    __shared__ __align__(16) float xs[64 * XS_STRIDE];
    const int tid = threadIdx.x;
    const int b       = blockIdx.y;
    const int tokbase = blockIdx.x * 64;
    const float* xb = x + (size_t)b * CIN * NTOK;
    const int stok = tid & 63;
    const int sc0  = (tid >> 6) * 8;
    const int wtok = tid >> 2;
    const int wseg = (tid & 3) * 8;

    for (int c0 = 0; c0 < CIN; c0 += 32) {
#pragma unroll
        for (int i = 0; i < 8; ++i) {
            const float* g = xb + (size_t)(c0 + sc0 + i) * NTOK + tokbase + stok;
            unsigned lds = (unsigned)(size_t)&xs[stok * XS_STRIDE + sc0 + i];
            async_copy_b32(lds, g);
        }
        wait_async0();
        __syncthreads();
        union { float4 v; float f[4]; } a0, a1;
        a0.v = *(const float4*)(const void*)&xs[wtok * XS_STRIDE + wseg];
        a1.v = *(const float4*)(const void*)&xs[wtok * XS_STRIDE + wseg + 4];
        union { uint4 u; unsigned short s[8]; } pk;
#pragma unroll
        for (int i = 0; i < 4; ++i) pk.s[i]     = bfbits(a0.f[i]);
#pragma unroll
        for (int i = 0; i < 4; ++i) pk.s[4 + i] = bfbits(a1.f[i]);
        bf16_t* dst = XB + ((size_t)b * NTOK + tokbase + wtok) * CIN + c0 + wseg;
        *(uint4*)(void*)dst = pk.u;
        __syncthreads();
    }
}

// ---------------- kernel 1: QKV GEMM (all-bf16, K-step 64) ----------------
__global__ __launch_bounds__(256) void qkv_kernel(const bf16_t* __restrict__ XB,
                                                  const bf16_t* __restrict__ wq,
                                                  const float* __restrict__ bias,
                                                  bf16_t* __restrict__ Q,
                                                  bf16_t* __restrict__ Kt,
                                                  bf16_t* __restrict__ VT) {
    __shared__ __align__(16) bf16_t bs[64 * BS_STRIDE];   // 9216 B
    const int tid  = threadIdx.x;
    const int lane = tid & 31;
    const int wave = tid >> 5;
    const int b       = blockIdx.z;
    const int obase   = blockIdx.y * 128 + wave * 16;
    const int tokbase = blockIdx.x * 64;
    const int grp = lane >> 4;
    const int n   = lane & 15;
    const int stok = tid >> 2;
    const int schk = (tid & 3) * 16;

    v8f acc[4];
#pragma unroll
    for (int t = 0; t < 4; ++t)
#pragma unroll
        for (int j = 0; j < 8; ++j) acc[t][j] = 0.0f;

    for (int k0 = 0; k0 < CIN; k0 += 64) {
        // stage 64tok x 64c bf16 tile (two b128 per thread)
        {
            const bf16_t* g = XB + ((size_t)b * NTOK + tokbase + stok) * CIN + k0 + schk;
            unsigned lds = (unsigned)(size_t)&bs[stok * BS_STRIDE + schk];
            async_copy_b128(lds, g);
            async_copy_b128(lds + 16, g + 8);
        }
        wait_async0();
        __syncthreads();

#pragma unroll
        for (int kk = 0; kk < 64; kk += 32) {
            v16bf a = load_a_bf16_w(wq, CIN, obase, k0 + kk, lane);
#pragma unroll
            for (int t = 0; t < 4; ++t) {
                const bf16_t* p = &bs[(t * 16 + n) * BS_STRIDE + kk + (grp << 4)];
                acc[t] = wmma_bf16(a, load_b_bf16(p), acc[t]);
            }
        }
        __syncthreads();
    }

    // epilogue: bias (+ folded softmax scale for Q) + scatter
    const int o0 = obase + grp * 8;
    const int region = o0 >> 9;                     // 0:Q 1:K 2:V
    const int rr   = o0 & 511;
    const int head = rr >> 6;
    const int d0   = rr & 63;
    const size_t bh = (size_t)b * HEADS + head;
#pragma unroll
    for (int t = 0; t < 4; ++t) {
        const int tok = tokbase + t * 16 + n;
        float vals[8];
#pragma unroll
        for (int j = 0; j < 8; ++j) {
            vals[j] = acc[t][j] + bias[o0 + j];
            if (region == 0) vals[j] *= QSCALE;
        }
        if (region < 2) {
            bf16_t* dst = (region == 0 ? Q : Kt) + (bh * NTOK + tok) * DH + d0;
            union { uint4 u; unsigned short s[8]; } pk;
#pragma unroll
            for (int j = 0; j < 8; ++j) pk.s[j] = bfbits(vals[j]);
            *(uint4*)(void*)dst = pk.u;
        } else {
            bf16_t* dst = VT + (bh * DH + d0) * NTOK + tok;
#pragma unroll
            for (int j = 0; j < 8; ++j) dst[(size_t)j * NTOK] = (bf16_t)vals[j];
        }
    }
}

// ---------------- kernel 2: causal flash attention (exp2 domain) ----------------
// Mask is applied only on the final (diagonal) 32-key tile.
#define ATTN_STEP(KB, MASKED)                                                        \
    do {                                                                             \
        const int kb_ = (KB);                                                        \
        v8f s[2];                                                                    \
        _Pragma("unroll")                                                            \
        for (int t = 0; t < 2; ++t) {                                                \
            _Pragma("unroll")                                                        \
            for (int j = 0; j < 8; ++j) s[t][j] = 0.0f;                              \
            const bf16_t* krow = Kb + (size_t)(kb_ + t * 16 + n) * DH + (grp << 4);  \
            v16bf b0 = load_b_bf16(krow);                                            \
            v16bf b1 = load_b_bf16(krow + 32);                                       \
            s[t] = wmma_bf16(qf0, b0, s[t]);                                         \
            s[t] = wmma_bf16(qf1, b1, s[t]);                                         \
        }                                                                            \
        if (MASKED) {                                                                \
            _Pragma("unroll")                                                        \
            for (int t = 0; t < 2; ++t)                                              \
                _Pragma("unroll")                                                    \
                for (int j = 0; j < 8; ++j) {                                        \
                    int kidx = kb_ + t * 16 + n;                                     \
                    int qidx = qtile + j + 8 * grp;                                  \
                    s[t][j] = (kidx <= qidx) ? s[t][j] : -3.0e38f;                   \
                }                                                                    \
        }                                                                            \
        float corr[8];                                                               \
        _Pragma("unroll")                                                            \
        for (int j = 0; j < 8; ++j) {                                                \
            float tm = fmaxf(s[0][j], s[1][j]);                                      \
            _Pragma("unroll")                                                        \
            for (int sh = 1; sh < 16; sh <<= 1)                                      \
                tm = fmaxf(tm, __shfl_xor(tm, sh, 32));                              \
            float mnew = fmaxf(mi[j], tm);                                           \
            float c  = fast_exp2(mi[j] - mnew);                                      \
            float p0 = fast_exp2(s[0][j] - mnew);                                    \
            float p1 = fast_exp2(s[1][j] - mnew);                                    \
            s[0][j] = p0; s[1][j] = p1;                                              \
            float rs = p0 + p1;                                                      \
            _Pragma("unroll")                                                        \
            for (int sh = 1; sh < 16; sh <<= 1)                                      \
                rs += __shfl_xor(rs, sh, 32);                                        \
            li[j] = li[j] * c + rs;                                                  \
            mi[j] = mnew;                                                            \
            corr[j] = c;                                                             \
        }                                                                            \
        _Pragma("unroll")                                                            \
        for (int t = 0; t < 4; ++t)                                                  \
            _Pragma("unroll")                                                        \
            for (int j = 0; j < 8; ++j) o[t][j] *= corr[j];                          \
        _Pragma("unroll")                                                            \
        for (int t = 0; t < 2; ++t)                                                  \
            _Pragma("unroll")                                                        \
            for (int j = 0; j < 8; ++j)                                              \
                pb[(j + 8 * grp) * 32 + t * 16 + n] = (bf16_t)s[t][j];               \
        asm volatile("s_wait_dscnt 0" ::: "memory");                                 \
        v16bf pa = load_a_bf16(pb + n * 32 + off);                                   \
        asm volatile("s_wait_dscnt 0" ::: "memory");                                 \
        _Pragma("unroll")                                                            \
        for (int t = 0; t < 4; ++t) {                                                \
            const bf16_t* vrow = Vb + (size_t)(t * 16 + n) * NTOK + kb_ + (grp << 4);\
            v16bf vb = load_b_bf16(vrow);                                            \
            o[t] = wmma_bf16(pa, vb, o[t]);                                          \
        }                                                                            \
    } while (0)

__global__ __launch_bounds__(256) void attn_kernel(const bf16_t* __restrict__ Q,
                                                   const bf16_t* __restrict__ Kt,
                                                   const bf16_t* __restrict__ VT,
                                                   bf16_t* __restrict__ AO) {
    __shared__ __align__(16) bf16_t pbuf[8][16 * 32];
    const int lane  = threadIdx.x & 31;
    const int wave  = threadIdx.x >> 5;
    const int bh    = blockIdx.y;
    const int qtile = blockIdx.x * 128 + wave * 16;

    const bf16_t* Qb = Q  + (size_t)bh * NTOK * DH;
    const bf16_t* Kb = Kt + (size_t)bh * NTOK * DH;
    const bf16_t* Vb = VT + (size_t)bh * DH * NTOK;

    const int grp = lane >> 4;
    const int n   = lane & 15;
    const int off = grp << 4;

    const bf16_t* qrow = Qb + (size_t)(qtile + n) * DH;
    v16bf qf0 = load_a_bf16(qrow + off);
    v16bf qf1 = load_a_bf16(qrow + 32 + off);

    float mi[8], li[8];
    v8f o[4];
#pragma unroll
    for (int j = 0; j < 8; ++j) { mi[j] = -3.0e38f; li[j] = 0.0f; }
#pragma unroll
    for (int t = 0; t < 4; ++t)
#pragma unroll
        for (int j = 0; j < 8; ++j) o[t][j] = 0.0f;

    bf16_t* pb = &pbuf[wave][0];

    const int kb_last = ((qtile + 15) >> 5) << 5;    // the only tile needing the mask
    for (int kb = 0; kb < kb_last; kb += 32) {
        __builtin_prefetch(Kb + (size_t)(kb + 32 + n) * DH, 0, 1);
        ATTN_STEP(kb, false);
    }
    ATTN_STEP(kb_last, true);

    // normalize (one reciprocal per row) + store AO[bh][tok][dv]
#pragma unroll
    for (int j = 0; j < 8; ++j) {
        float inv = 1.0f / li[j];
        int tok = qtile + j + 8 * grp;
#pragma unroll
        for (int t = 0; t < 4; ++t) {
            int dv = t * 16 + n;
            AO[((size_t)bh * NTOK + tok) * DH + dv] = (bf16_t)(o[t][j] * inv);
        }
    }
}

// ---------------- kernel 3: output projection GEMM (all-bf16, K-step 64) ----------------
__global__ __launch_bounds__(256) void proj_kernel(const bf16_t* __restrict__ AO,
                                                   const bf16_t* __restrict__ wp,
                                                   const float* __restrict__ bias,
                                                   float* __restrict__ out) {
    __shared__ __align__(16) bf16_t bs[64 * BS_STRIDE];
    const int tid  = threadIdx.x;
    const int lane = tid & 31;
    const int wave = tid >> 5;
    const int b       = blockIdx.z;
    const int obase   = blockIdx.y * 128 + wave * 16;
    const int tokbase = blockIdx.x * 64;
    const int grp = lane >> 4;
    const int n   = lane & 15;
    const int stok = tid >> 2;
    const int schk = (tid & 3) * 16;

    v8f acc[4];
#pragma unroll
    for (int t = 0; t < 4; ++t)
#pragma unroll
        for (int j = 0; j < 8; ++j) acc[t][j] = 0.0f;

    for (int k0 = 0; k0 < VDIM; k0 += 64) {
        const int head = k0 >> 6;     // 64 v's == exactly one head
        {
            const bf16_t* g = AO + (((size_t)b * HEADS + head) * NTOK + tokbase + stok) * DH
                              + schk;
            unsigned lds = (unsigned)(size_t)&bs[stok * BS_STRIDE + schk];
            async_copy_b128(lds, g);
            async_copy_b128(lds + 16, g + 8);
        }
        wait_async0();
        __syncthreads();

#pragma unroll
        for (int kk = 0; kk < 64; kk += 32) {
            v16bf a = load_a_bf16_w(wp, VDIM, obase, k0 + kk, lane);
#pragma unroll
            for (int t = 0; t < 4; ++t) {
                const bf16_t* p = &bs[(t * 16 + n) * BS_STRIDE + kk + (grp << 4)];
                acc[t] = wmma_bf16(a, load_b_bf16(p), acc[t]);
            }
        }
        __syncthreads();
    }

#pragma unroll
    for (int t = 0; t < 4; ++t) {
        int tok = tokbase + t * 16 + n;
#pragma unroll
        for (int j = 0; j < 8; ++j) {
            int oo = obase + j + 8 * grp;
            out[((size_t)b * OUTCH + oo) * NTOK + tok] = acc[t][j] + bias[oo];
        }
    }
}

// ---------------- host side ----------------
extern "C" void kernel_launch(void* const* d_in, const int* in_sizes, int n_in,
                              void* d_out, int out_size, void* d_ws, size_t ws_size,
                              hipStream_t stream) {
    (void)in_sizes; (void)n_in; (void)out_size; (void)ws_size;
    const float* x      = (const float*)d_in[0];
    const float* qkv_w  = (const float*)d_in[1];
    const float* qkv_b  = (const float*)d_in[2];
    const float* proj_w = (const float*)d_in[3];
    const float* proj_b = (const float*)d_in[4];
    float* out = (float*)d_out;

    const size_t elems = (size_t)NBATCH * HEADS * NTOK * DH;       // 8 Mi
    bf16_t* Q  = (bf16_t*)d_ws;
    bf16_t* Kt = Q  + elems;
    bf16_t* VT = Kt + elems;
    bf16_t* AO = VT + elems;
    bf16_t* XB = AO + elems;                                       // 4 Mi
    bf16_t* WQ = XB + (size_t)NBATCH * NTOK * CIN;                 // 384 Ki
    bf16_t* WP = WQ + (size_t)QKVO * CIN;                          // 128 Ki

    cvt_kernel<<<dim3((QKVO * CIN) / 2048), 256, 0, stream>>>(qkv_w, WQ, QKVO * CIN);
    cvt_kernel<<<dim3((OUTCH * VDIM) / 2048), 256, 0, stream>>>(proj_w, WP, OUTCH * VDIM);
    xprep_kernel<<<dim3(NTOK / 64, NBATCH), 256, 0, stream>>>(x, XB);
    qkv_kernel<<<dim3(NTOK / 64, QKVO / 128, NBATCH), 256, 0, stream>>>(
        XB, WQ, qkv_b, Q, Kt, VT);
    attn_kernel<<<dim3(NTOK / 128, NBATCH * HEADS), 256, 0, stream>>>(
        Q, Kt, VT, AO);
    proj_kernel<<<dim3(NTOK / 64, OUTCH / 128, NBATCH), 256, 0, stream>>>(
        AO, WP, proj_b, out);
}
